// CustomModel_8770323218904
// MI455X (gfx1250) — matrix operand order repaired and checked by reading, not compiled, not run on hardware
//
#include <hip/hip_runtime.h>

typedef __attribute__((ext_vector_type(16))) _Float16 v16h;
typedef __attribute__((ext_vector_type(8)))  _Float16 v8h;
typedef __attribute__((ext_vector_type(8)))  float    v8f;

#define NB 2048  // batch

static __device__ __forceinline__ v16h cat8(v8h lo, v8h hi) {
  return __builtin_shufflevector(lo, hi, 0,1,2,3,4,5,6,7,8,9,10,11,12,13,14,15);
}
static __device__ __forceinline__ v8f wmma16x16x32(v16h a, v16h b, v8f c) {
  // 8 args: (neg_a, A, neg_b, B, c_mod, C, reuse_a, reuse_b)
  return __builtin_amdgcn_wmma_f32_16x16x32_f16(false, a, false, b, (short)0, c,
                                                false, false);
}

// ---------------------------------------------------------------------------
// Kernel 0: weight transforms f32 -> f16, laid out as [kstep][n][32] so each
// lane's B-fragment (col n, K [16h,16h+16)) is two contiguous b128 loads at
// compile-time immediate offsets.
// ---------------------------------------------------------------------------
#define W2T_N (9*64*32)
#define W3T_N (18*64*32)
#define WD1T_N (64*3136)
__global__ void prep_weights_k(const float* __restrict__ w2,
                               const float* __restrict__ w3,
                               const float* __restrict__ wd1,
                               _Float16* __restrict__ w2t,
                               _Float16* __restrict__ w3t,
                               _Float16* __restrict__ wd1t) {
  int idx = blockIdx.x * blockDim.x + threadIdx.x;
  if (idx < W2T_N) {
    int s = idx / (64*32), r = idx % (64*32), n = r / 32, kk = r % 32;
    w2t[idx] = (_Float16)w2[(s*32 + kk)*64 + n];
  } else if (idx < W2T_N + W3T_N) {
    int j = idx - W2T_N;
    int s = j / (64*32), r = j % (64*32), n = r / 32, kk = r % 32;
    w3t[j] = (_Float16)w3[(((s >> 1)*64) + ((s & 1)*32) + kk)*64 + n];
  } else if (idx < W2T_N + W3T_N + WD1T_N) {
    int j = idx - W2T_N - W3T_N;
    int n = j / 3136, k = j % 3136;
    wd1t[j] = (_Float16)wd1[(size_t)k*64 + n];
  }
}

// ---------------------------------------------------------------------------
// Kernel 1: conv1 (1->32ch, 3x3 SAME) + bias + relu + 2x2 maxpool, written
// into ZERO-PADDED h1p [B,16,16,32] f16 (border = 0) so conv2's halo taps
// never need predication. K=9 is too small for WMMA; VALU is right here.
// ---------------------------------------------------------------------------
__global__ __launch_bounds__(256) void conv1_pool_k(
    const float* __restrict__ x, const float* __restrict__ w1,
    const float* __restrict__ b1, _Float16* __restrict__ h1p) {
  int idx = blockIdx.x * blockDim.x + threadIdx.x;
  int c = idx & 31;
  int p = idx >> 5;                 // (b, py16, px16) padded coords
  if (p >= NB * 256) return;
  int b = p >> 8, pos = p & 255, py = pos >> 4, px = pos & 15;
  if (py == 0 || py == 15 || px == 0 || px == 15) {   // zero border
    h1p[(size_t)p*32 + c] = (_Float16)0.0f;
    return;
  }
  int oy0 = 2*(py - 1), ox0 = 2*(px - 1);             // 14x14 interior
  float wv[9];
#pragma unroll
  for (int k = 0; k < 9; ++k) wv[k] = w1[k*32 + c];
  const float bias = b1[c];
  float mx = 0.0f;                  // relu(.)>=0 so 0 is the pooling identity
#pragma unroll
  for (int i = 0; i < 2; ++i) {
#pragma unroll
    for (int j = 0; j < 2; ++j) {
      int oy = oy0 + i, ox = ox0 + j;
      float s = bias;
#pragma unroll
      for (int dy = 0; dy < 3; ++dy) {
#pragma unroll
        for (int dx = 0; dx < 3; ++dx) {
          int iy = oy + dy - 1, ix = ox + dx - 1;
          if ((unsigned)iy < 28u && (unsigned)ix < 28u)
            s += x[((size_t)b*28 + iy)*28 + ix] * wv[dy*3 + dx];
        }
      }
      s = s > 0.0f ? s : 0.0f;
      mx = fmaxf(mx, s);
    }
  }
  h1p[(size_t)p*32 + c] = (_Float16)mx;
}

// ---------------------------------------------------------------------------
// Kernel 2: conv2 implicit GEMM: M=B*196, N=64, K=288 (9 ksteps of 32).
// One wave -> full 16x64 output strip: 4 accumulators share each A fragment
// (4 WMMA per 2 A-loads). All tap/weight addresses are a single per-lane
// base + compile-time immediates (padded input => no predication, no EXEC
// churn, loads clause together). Output: h2 [B,14,14,64] f16 (pre-pool).
// ---------------------------------------------------------------------------
__global__ __launch_bounds__(256) void conv2_wmma_k(
    const _Float16* __restrict__ h1p, const _Float16* __restrict__ w2t,
    const float* __restrict__ b2, _Float16* __restrict__ h2) {
  const int lane  = threadIdx.x & 31;
  const int mtile = blockIdx.x * (blockDim.x >> 5) + (threadIdx.x >> 5);
  if (mtile >= (NB * 196) / 16) return;
  const int half = lane >> 4;
  const int l16  = lane & 15;
  const int mrow = (mtile << 4) + l16;           // A row owned by this lane
  const int b = mrow / 196, pos = mrow % 196;
  const int yy = pos / 14, xx = pos % 14;
  // Center-tap base in padded [B,16,16,32]; taps are +/- immediates.
  const _Float16* ap = h1p + (((size_t)(b << 8) + ((yy + 1) << 4) + (xx + 1)) << 5)
                           + (half << 4);
  const _Float16* bp = w2t + (l16 << 5) + (half << 4);

  v8f acc0 = {}, acc1 = {}, acc2 = {}, acc3 = {};
#pragma unroll
  for (int s = 0; s < 9; ++s) {
    const int aoff = (s/3 - 1)*512 + (s%3 - 1)*32;          // elements
    v16h a = cat8(*(const v8h*)(ap + aoff), *(const v8h*)(ap + aoff + 8));
    const int bo = s*2048;                                   // [s][n][32]
    v16h b0 = cat8(*(const v8h*)(bp + bo),        *(const v8h*)(bp + bo + 8));
    v16h b1 = cat8(*(const v8h*)(bp + bo + 512),  *(const v8h*)(bp + bo + 520));
    v16h b2m= cat8(*(const v8h*)(bp + bo + 1024), *(const v8h*)(bp + bo + 1032));
    v16h b3 = cat8(*(const v8h*)(bp + bo + 1536), *(const v8h*)(bp + bo + 1544));
    acc0 = wmma16x16x32(a, b0, acc0);
    acc1 = wmma16x16x32(a, b1, acc1);
    acc2 = wmma16x16x32(a, b2m, acc2);
    acc3 = wmma16x16x32(a, b3, acc3);
  }
  const float bi0 = b2[l16], bi1 = b2[16 + l16], bi2 = b2[32 + l16], bi3 = b2[48 + l16];
#pragma unroll
  for (int r = 0; r < 8; ++r) {                  // D: lane holds (M=r+8h, N)
    size_t row = ((size_t)(mtile << 4) + r + (half << 3)) * 64 + l16;
    float v0 = acc0[r] + bi0; v0 = v0 > 0.f ? v0 : 0.f;
    float v1 = acc1[r] + bi1; v1 = v1 > 0.f ? v1 : 0.f;
    float v2 = acc2[r] + bi2; v2 = v2 > 0.f ? v2 : 0.f;
    float v3 = acc3[r] + bi3; v3 = v3 > 0.f ? v3 : 0.f;
    h2[row]      = (_Float16)v0;
    h2[row + 16] = (_Float16)v1;
    h2[row + 32] = (_Float16)v2;
    h2[row + 48] = (_Float16)v3;
  }
}

// ---------------------------------------------------------------------------
// Kernel 3: 2x2 maxpool h2 -> ZERO-PADDED h2p [B,9,9,64] f16 (border = 0).
// ---------------------------------------------------------------------------
__global__ __launch_bounds__(256) void maxpool2_k(
    const _Float16* __restrict__ h2, _Float16* __restrict__ h2p) {
  int idx = blockIdx.x * blockDim.x + threadIdx.x;
  int c = idx & 63;
  int p = idx >> 6;                               // (b, py9, px9)
  if (p >= NB * 81) return;
  int b = p / 81, pos = p % 81, py = pos / 9, px = pos % 9;
  if (py == 0 || py == 8 || px == 0 || px == 8) {
    h2p[(size_t)p*64 + c] = (_Float16)0.0f;
    return;
  }
  size_t base = (((size_t)b*14 + 2*(py-1))*14 + 2*(px-1))*64 + c;
  float m0 = fmaxf((float)h2[base],         (float)h2[base + 64]);
  float m1 = fmaxf((float)h2[base + 14*64], (float)h2[base + 14*64 + 64]);
  h2p[(size_t)p*64 + c] = (_Float16)fmaxf(m0, m1);
}

// ---------------------------------------------------------------------------
// Kernel 4: conv3 implicit GEMM: M=B*49, N=64, K=576 (18 ksteps: tap=s>>1,
// channel-half=(s&1)*32). Padded input, immediate-offset taps, 4 acc/wave.
// Output h3 [B,7,7,64] f16 UNPADDED (== keras Flatten order for dense1).
// ---------------------------------------------------------------------------
__global__ __launch_bounds__(256) void conv3_wmma_k(
    const _Float16* __restrict__ h2p, const _Float16* __restrict__ w3t,
    const float* __restrict__ b3, _Float16* __restrict__ h3) {
  const int lane  = threadIdx.x & 31;
  const int mtile = blockIdx.x * (blockDim.x >> 5) + (threadIdx.x >> 5);
  if (mtile >= (NB * 49) / 16) return;
  const int half = lane >> 4;
  const int l16  = lane & 15;
  const int mrow = (mtile << 4) + l16;
  const int b = mrow / 49, pos = mrow % 49;
  const int yy = pos / 7, xx = pos % 7;
  // Center-tap base in padded [B,9,9,64].
  const _Float16* ap = h2p + (((size_t)b*81 + (yy + 1)*9 + (xx + 1)) << 6)
                           + (half << 4);
  const _Float16* bp = w3t + (l16 << 5) + (half << 4);

  v8f acc0 = {}, acc1 = {}, acc2 = {}, acc3 = {};
#pragma unroll
  for (int s = 0; s < 18; ++s) {
    const int tap = s >> 1;
    const int aoff = (tap/3 - 1)*576 + (tap%3 - 1)*64 + ((s & 1) << 5); // elems
    v16h a = cat8(*(const v8h*)(ap + aoff), *(const v8h*)(ap + aoff + 8));
    const int bo = s*2048;
    v16h b0 = cat8(*(const v8h*)(bp + bo),        *(const v8h*)(bp + bo + 8));
    v16h b1 = cat8(*(const v8h*)(bp + bo + 512),  *(const v8h*)(bp + bo + 520));
    v16h b2m= cat8(*(const v8h*)(bp + bo + 1024), *(const v8h*)(bp + bo + 1032));
    v16h b3v= cat8(*(const v8h*)(bp + bo + 1536), *(const v8h*)(bp + bo + 1544));
    acc0 = wmma16x16x32(a, b0, acc0);
    acc1 = wmma16x16x32(a, b1, acc1);
    acc2 = wmma16x16x32(a, b2m, acc2);
    acc3 = wmma16x16x32(a, b3v, acc3);
  }
  const float bi0 = b3[l16], bi1 = b3[16 + l16], bi2 = b3[32 + l16], bi3 = b3[48 + l16];
#pragma unroll
  for (int r = 0; r < 8; ++r) {
    size_t row = ((size_t)(mtile << 4) + r + (half << 3)) * 64 + l16;
    float v0 = acc0[r] + bi0; v0 = v0 > 0.f ? v0 : 0.f;
    float v1 = acc1[r] + bi1; v1 = v1 > 0.f ? v1 : 0.f;
    float v2 = acc2[r] + bi2; v2 = v2 > 0.f ? v2 : 0.f;
    float v3 = acc3[r] + bi3; v3 = v3 > 0.f ? v3 : 0.f;
    h3[row]      = (_Float16)v0;
    h3[row + 16] = (_Float16)v1;
    h3[row + 32] = (_Float16)v2;
    h3[row + 48] = (_Float16)v3;
  }
}

// ---------------------------------------------------------------------------
// Kernel 5: dense1 GEMM: M=2048, K=3136 (98 ksteps), N=64 (4 acc per wave).
// Fully contiguous A/B; pointer-bumped loop, N-tiles at immediate offsets.
// ---------------------------------------------------------------------------
__global__ __launch_bounds__(256) void dense1_wmma_k(
    const _Float16* __restrict__ h3, const _Float16* __restrict__ wd1t,
    const float* __restrict__ bd1, _Float16* __restrict__ h4) {
  const int lane  = threadIdx.x & 31;
  const int mtile = blockIdx.x * (blockDim.x >> 5) + (threadIdx.x >> 5);
  if (mtile >= NB / 16) return;
  const int half = lane >> 4;
  const int l16  = lane & 15;
  const int mrow = (mtile << 4) + l16;
  const _Float16* ap = h3 + (size_t)mrow * 3136 + (half << 4);
  const _Float16* bp = wd1t + (size_t)l16 * 3136 + (half << 4);

  v8f acc0 = {}, acc1 = {}, acc2 = {}, acc3 = {};
#pragma unroll 2
  for (int k = 0; k < 3136; k += 32) {
    v16h a = cat8(*(const v8h*)ap, *(const v8h*)(ap + 8));
    v16h b0 = cat8(*(const v8h*)bp,             *(const v8h*)(bp + 8));
    v16h b1 = cat8(*(const v8h*)(bp + 16*3136), *(const v8h*)(bp + 16*3136 + 8));
    v16h b2 = cat8(*(const v8h*)(bp + 32*3136), *(const v8h*)(bp + 32*3136 + 8));
    v16h b3 = cat8(*(const v8h*)(bp + 48*3136), *(const v8h*)(bp + 48*3136 + 8));
    acc0 = wmma16x16x32(a, b0, acc0);
    acc1 = wmma16x16x32(a, b1, acc1);
    acc2 = wmma16x16x32(a, b2, acc2);
    acc3 = wmma16x16x32(a, b3, acc3);
    ap += 32; bp += 32;
  }
  const float bi0 = bd1[l16], bi1 = bd1[16+l16], bi2 = bd1[32+l16], bi3 = bd1[48+l16];
#pragma unroll
  for (int r = 0; r < 8; ++r) {
    size_t row = ((size_t)(mtile << 4) + r + (half << 3)) * 64 + l16;
    float v0 = acc0[r] + bi0; v0 = v0 > 0.f ? v0 : 0.f;
    float v1 = acc1[r] + bi1; v1 = v1 > 0.f ? v1 : 0.f;
    float v2 = acc2[r] + bi2; v2 = v2 > 0.f ? v2 : 0.f;
    float v3 = acc3[r] + bi3; v3 = v3 > 0.f ? v3 : 0.f;
    h4[row]      = (_Float16)v0;
    h4[row + 16] = (_Float16)v1;
    h4[row + 32] = (_Float16)v2;
    h4[row + 48] = (_Float16)v3;
  }
}

// ---------------------------------------------------------------------------
// Kernel 6: dense2 (64->10) + numerically-stable softmax, f32 out.
// ---------------------------------------------------------------------------
__global__ __launch_bounds__(256) void dense2_softmax_k(
    const _Float16* __restrict__ h4, const float* __restrict__ wd2,
    const float* __restrict__ bd2, float* __restrict__ out) {
  int b = blockIdx.x * blockDim.x + threadIdx.x;
  if (b >= NB) return;
  float logits[10];
#pragma unroll
  for (int n = 0; n < 10; ++n) logits[n] = bd2[n];
  for (int k = 0; k < 64; ++k) {
    float hv = (float)h4[(size_t)b*64 + k];
#pragma unroll
    for (int n = 0; n < 10; ++n) logits[n] += hv * wd2[k*10 + n];
  }
  float mx = logits[0];
#pragma unroll
  for (int n = 1; n < 10; ++n) mx = fmaxf(mx, logits[n]);
  float sum = 0.0f;
#pragma unroll
  for (int n = 0; n < 10; ++n) { logits[n] = __expf(logits[n] - mx); sum += logits[n]; }
  float inv = 1.0f / sum;
#pragma unroll
  for (int n = 0; n < 10; ++n) out[(size_t)b*10 + n] = logits[n] * inv;
}

// ---------------------------------------------------------------------------
extern "C" void kernel_launch(void* const* d_in, const int* in_sizes, int n_in,
                              void* d_out, int out_size, void* d_ws, size_t ws_size,
                              hipStream_t stream) {
  (void)in_sizes; (void)n_in; (void)out_size; (void)ws_size;
  const float* x   = (const float*)d_in[0];
  const float* w1  = (const float*)d_in[1];
  const float* b1  = (const float*)d_in[2];
  const float* w2  = (const float*)d_in[3];
  const float* b2  = (const float*)d_in[4];
  const float* w3  = (const float*)d_in[5];
  const float* b3  = (const float*)d_in[6];
  const float* wd1 = (const float*)d_in[7];
  const float* bd1 = (const float*)d_in[8];
  const float* wd2 = (const float*)d_in[9];
  const float* bd2 = (const float*)d_in[10];
  float* out = (float*)d_out;

  // Workspace carve-up (all 256B aligned).
  char* ws = (char*)d_ws;
  size_t off = 0;
  _Float16* h1p  = (_Float16*)(ws + off); off += (size_t)NB*256*32*2;  // 33.6 MB padded
  _Float16* h2   = (_Float16*)(ws + off); off += (size_t)NB*196*64*2;  // 51.4 MB
  _Float16* h2p  = (_Float16*)(ws + off); off += (size_t)NB*81*64*2;   // 21.2 MB padded
  _Float16* h3   = (_Float16*)(ws + off); off += (size_t)NB*49*64*2;   // 12.8 MB
  _Float16* h4   = (_Float16*)(ws + off); off += (size_t)NB*64*2;      // 0.26 MB
  _Float16* w2t  = (_Float16*)(ws + off); off += (size_t)W2T_N*2;
  _Float16* w3t  = (_Float16*)(ws + off); off += (size_t)W3T_N*2;
  _Float16* wd1t = (_Float16*)(ws + off); off += (size_t)WD1T_N*2;

  const int T = 256;
  // 0) weight transforms
  prep_weights_k<<<(W2T_N + W3T_N + WD1T_N + T - 1)/T, T, 0, stream>>>(
      w2, w3, wd1, w2t, w3t, wd1t);
  // 1) conv1 + relu + pool into padded h1p (B*256*32 threads incl. border)
  conv1_pool_k<<<(NB*256*32 + T - 1)/T, T, 0, stream>>>(x, w1, b1, h1p);
  // 2) conv2 WMMA: one wave per 16x64 strip -> (B*196/16) waves, 8/block
  conv2_wmma_k<<<((NB*196/16) + 7)/8, T, 0, stream>>>(h1p, w2t, b2, h2);
  // 3) maxpool 2x2 into padded h2p (incl. border)
  maxpool2_k<<<(NB*81*64 + T - 1)/T, T, 0, stream>>>(h2, h2p);
  // 4) conv3 WMMA
  conv3_wmma_k<<<((NB*49/16) + 7)/8, T, 0, stream>>>(h2p, w3t, b3, h3);
  // 5) dense1 WMMA
  dense1_wmma_k<<<((NB/16) + 7)/8, T, 0, stream>>>(h3, wd1t, bd1, h4);
  // 6) dense2 + softmax
  dense2_softmax_k<<<(NB + T - 1)/T, T, 0, stream>>>(h4, wd2, bd2, out);
}
